// MultiHeadCDSA_24154896073398
// MI455X (gfx1250) — matrix operand order repaired and checked by reading, not compiled, not run on hardware
//
#include <hip/hip_runtime.h>
#include <hip/hip_bf16.h>

// ---------------------------------------------------------------------------
// MultiHeadCDSA for MI455X (gfx1250): all matmuls via v_wmma_f32_16x16x32_f16.
// B=4, N=1024, D=1024, H=8, DK=DV=128.  Every GEMM has M % 128 == 0,
// N % 64 == 0, K % 32 == 0, so no edge predication (EXEC all-ones for WMMA).
// Block tile 128x64x32, 8 wave32s in a 4(M) x 2(N) grid, each wave computes a
// 32x32 C tile = four 16x16x32 WMMA accumulators.  LDS double-buffered; all
// fragment ds_loads hoisted ahead of the 4 back-to-back WMMAs.
// ---------------------------------------------------------------------------

typedef __attribute__((ext_vector_type(16))) _Float16 v16h;
typedef __attribute__((ext_vector_type(8)))  _Float16 v8h;
typedef __attribute__((ext_vector_type(8)))  float    v8f;

#define Bc  4
#define Nc  1024
#define Dc  1024
#define Hc  8
#define DKc 128
#define DVc 128

struct GemmP {
    const void* A;
    const void* Bm;
    void*       C;          // f16 (MODE 0) or f32 (MODE 1/2)
    const float* aux;       // residual for MODE 2 (same indexing as C)
    long long a_sb, a_sh;   // batch offsets: off = (bz/HH)*sb + (bz%HH)*sh
    long long b_sb, b_sh;
    long long c_sb, c_sh;
    int lda_m, lda_k;       // addrA(m,k) = aOff + m*lda_m + k*lda_k
    int ldb_k, ldb_n;       // addrB(k,n) = bOff + k*ldb_k + n*ldb_n
    int ldc;                // addrC(m,n) = cOff + m*ldc + n
    int K;
    int HH;                 // batch decomposition divisor (H or 1)
    float alpha;            // scale applied to accumulator before store
};

struct ARegs { v8h x, y; };   // 16 staged A halves per thread

// AM: 0 = f16 k-fast (lda_k==1), 1 = f32 k-fast, 2 = f16 m-fast (lda_m==1)
// BM: 0 = f16 k-fast (ldb_k==1), 1 = f32 k-fast, 2 = f16 n-fast (ldb_n==1),
//     3 = f32 n-fast
// MODE: 0 = f16 store, 1 = f32 store, 2 = f32 + residual
template<int AM, int BM, int MODE>
__global__ __launch_bounds__(256)
void gemm_wmma(GemmP p) {
    __shared__ _Float16 As[2][128 * 40];  // [m][k], padded row stride 40 halves
    __shared__ _Float16 Bs[2][64 * 40];   // [n][k] (transposed), same padding

    const int tid = threadIdx.x;
    const int bz  = blockIdx.z;
    const int bb  = bz / p.HH;
    const int hh  = bz % p.HH;
    const long long aOff = (long long)bb * p.a_sb + (long long)hh * p.a_sh;
    const long long bOff = (long long)bb * p.b_sb + (long long)hh * p.b_sh;
    const long long cOff = (long long)bb * p.c_sb + (long long)hh * p.c_sh;

    const long long rowM = (long long)blockIdx.y * 128;
    const long long colN = (long long)blockIdx.x * 64;

    const float*    Af = (const float*)p.A;
    const _Float16* Ah = (const _Float16*)p.A;
    const float*    Bf = (const float*)p.Bm;
    const _Float16* Bh = (const _Float16*)p.Bm;

    // A staging (128x32 tile, 16 elems/thread):
    //   k-fast:  row rA = tid>>1, 16 consecutive k at kA = (tid&1)*16
    //   m-fast:  col kSC = tid>>3, 16 consecutive m at mA2 = (tid&7)*16
    const int rA  = tid >> 1;
    const int kA  = (tid & 1) * 16;
    const int kSC = tid >> 3;            // 0..31
    const int mA2 = (tid & 7) * 16;      // 0..112
    // B staging (32x64 tile, 8 elems/thread):
    //   k-fast:  col nB = tid>>2, 8 consecutive k at kB = (tid&3)*8
    //   n-fast:  row kSC, 8 consecutive n at nSC = (tid&7)*8
    const int nB  = tid >> 2;            // 0..63
    const int kB  = (tid & 3) * 8;       // 0,8,16,24
    const int nSC = (tid & 7) * 8;       // 0..56

    auto cvt8 = [](float4 f0, float4 f1) {
        v8h r;
        r[0] = (_Float16)f0.x; r[1] = (_Float16)f0.y;
        r[2] = (_Float16)f0.z; r[3] = (_Float16)f0.w;
        r[4] = (_Float16)f1.x; r[5] = (_Float16)f1.y;
        r[6] = (_Float16)f1.z; r[7] = (_Float16)f1.w;
        return r;
    };

    auto loadA = [&](int k0) -> ARegs {
        ARegs r;
        if constexpr (AM == 0) {
            const long long g = aOff + (rowM + rA) * (long long)p.lda_m + (k0 + kA);
            r.x = *(const v8h*)(Ah + g);
            r.y = *(const v8h*)(Ah + g + 8);
        } else if constexpr (AM == 1) {
            const long long g = aOff + (rowM + rA) * (long long)p.lda_m + (k0 + kA);
            r.x = cvt8(*(const float4*)(Af + g),     *(const float4*)(Af + g + 4));
            r.y = cvt8(*(const float4*)(Af + g + 8), *(const float4*)(Af + g + 12));
        } else {            // AM==2: m-fast (transposed A view)
            const long long g = aOff + (rowM + mA2) + (long long)(k0 + kSC) * p.lda_k;
            r.x = *(const v8h*)(Ah + g);
            r.y = *(const v8h*)(Ah + g + 8);
        }
        return r;
    };
    auto storeA = [&](_Float16* dst, ARegs r) {
        if constexpr (AM == 2) {
#pragma unroll
            for (int i = 0; i < 8; ++i) {
                dst[(mA2 + i)     * 40 + kSC] = r.x[i];
                dst[(mA2 + 8 + i) * 40 + kSC] = r.y[i];
            }
        } else {
            *(v8h*)(dst + rA * 40 + kA)     = r.x;
            *(v8h*)(dst + rA * 40 + kA + 8) = r.y;
        }
    };

    auto loadB = [&](int k0) -> v8h {
        if constexpr (BM == 0) {
            const long long g = bOff + (k0 + kB) + (colN + nB) * (long long)p.ldb_n;
            return *(const v8h*)(Bh + g);
        } else if constexpr (BM == 1) {
            const long long g = bOff + (k0 + kB) + (colN + nB) * (long long)p.ldb_n;
            return cvt8(*(const float4*)(Bf + g), *(const float4*)(Bf + g + 4));
        } else if constexpr (BM == 2) {   // f16 n-fast
            const long long g = bOff + (long long)(k0 + kSC) * p.ldb_k + (colN + nSC);
            return *(const v8h*)(Bh + g);
        } else {                          // f32 n-fast
            const long long g = bOff + (long long)(k0 + kSC) * p.ldb_k + (colN + nSC);
            return cvt8(*(const float4*)(Bf + g), *(const float4*)(Bf + g + 4));
        }
    };
    auto storeB = [&](_Float16* dst, v8h r) {
        if constexpr (BM >= 2) {
#pragma unroll
            for (int i = 0; i < 8; ++i) dst[(nSC + i) * 40 + kSC] = r[i];
        } else {
            *(v8h*)(dst + nB * 40 + kB) = r;
        }
    };

    const int wave = tid >> 5;
    const int lane = tid & 31;
    const int wm   = wave >> 1;          // 0..3 -> M offset wm*32
    const int wn   = wave & 1;           // 0..1 -> N offset wn*32
    const int l16  = lane & 15;
    const int hi   = (lane >> 4) & 1;

    v8f acc[2][2];
#pragma unroll
    for (int mi = 0; mi < 2; ++mi)
#pragma unroll
        for (int ni = 0; ni < 2; ++ni)
#pragma unroll
            for (int i = 0; i < 8; ++i) acc[mi][ni][i] = 0.0f;

    const int nK = p.K >> 5;

    // prologue: stage tile 0
    {
        ARegs a0 = loadA(0); v8h b0 = loadB(0);
        storeA(As[0], a0);
        storeB(Bs[0], b0);
    }
    __syncthreads();

    for (int kt = 0; kt < nK; ++kt) {
        const int cur = kt & 1, nxt = cur ^ 1;

        // issue next tile's global loads before consuming current tile
        ARegs aReg; v8h bReg;
        if (kt + 1 < nK) {
            aReg = loadA((kt + 1) << 5);
            bReg = loadB((kt + 1) << 5);
        }

        // ---- all fragment ds_loads first, then 4 back-to-back WMMAs -------
        const _Float16* Ac = As[cur];
        const _Float16* Bcur = Bs[cur];
        const int ak = hi ? 8 : 0;       // A: K halves {ak..ak+7, ak+16..ak+23}
        const int bk = hi ? 8 : 0;       // B: K halves {bk..bk+7, bk+16..bk+23}

        v16h afrag[2], bfrag[2];
#pragma unroll
        for (int mi = 0; mi < 2; ++mi) {
            const int am = wm * 32 + mi * 16 + l16;
            const v8h alo = *(const v8h*)&Ac[am * 40 + ak];
            const v8h ahi = *(const v8h*)&Ac[am * 40 + ak + 16];
#pragma unroll
            for (int i = 0; i < 8; ++i) { afrag[mi][i] = alo[i]; afrag[mi][i + 8] = ahi[i]; }
        }
#pragma unroll
        for (int ni = 0; ni < 2; ++ni) {
            const int bn = wn * 32 + ni * 16 + l16;
            const v8h b0 = *(const v8h*)&Bcur[bn * 40 + bk];
            const v8h b1 = *(const v8h*)&Bcur[bn * 40 + bk + 16];
#pragma unroll
            for (int i = 0; i < 8; ++i) { bfrag[ni][i] = b0[i]; bfrag[ni][i + 8] = b1[i]; }
        }
#pragma unroll
        for (int mi = 0; mi < 2; ++mi)
#pragma unroll
            for (int ni = 0; ni < 2; ++ni)
                acc[mi][ni] = __builtin_amdgcn_wmma_f32_16x16x32_f16(
                    false, afrag[mi], false, bfrag[ni], (short)0, acc[mi][ni],
                    false, false);

        if (kt + 1 < nK) {
            storeA(As[nxt], aReg);
            storeB(Bs[nxt], bReg);
        }
        __syncthreads();
    }

    // ---- epilogue: D layout VGPR j -> M=(hi?8:0)+j, N=l16 -----------------
    _Float16* Ch = (_Float16*)p.C;
    float*    Cf = (float*)p.C;
#pragma unroll
    for (int mi = 0; mi < 2; ++mi) {
        const long long row0 = rowM + wm * 32 + mi * 16 + hi * 8;
#pragma unroll
        for (int ni = 0; ni < 2; ++ni) {
            const long long col = colN + wn * 32 + ni * 16 + l16;
#pragma unroll
            for (int j = 0; j < 8; ++j) {
                const float r = acc[mi][ni][j] * p.alpha;
                const long long ci = cOff + (row0 + j) * (long long)p.ldc + col;
                if constexpr (MODE == 0)      Ch[ci] = (_Float16)r;
                else if constexpr (MODE == 1) Cf[ci] = r;
                else Cf[ci] = r + p.aux[(row0 + j) * (long long)p.ldc + col];
            }
        }
    }
}

// ---- masked softmax over rows of length 1024 (attn_time), in place --------
__global__ __launch_bounds__(256)
void softmax_time(float* at, const int* mask) {
    const int row = blockIdx.x;
    const int bz  = blockIdx.y;
    const int b   = bz / Hc;
    float* p      = at   + ((long long)bz * Nc + row) * Nc;
    const int* mp = mask + ((long long)b  * Nc + row) * Nc;

    __shared__ float red[256];
    float v[4];
    float mx = -3.4e38f;
#pragma unroll
    for (int i = 0; i < 4; ++i) {
        const int c = threadIdx.x + i * 256;
        float x = p[c];
        if (mp[c] == 0) x = -1e9f;
        v[i] = x;
        mx = fmaxf(mx, x);
    }
    red[threadIdx.x] = mx; __syncthreads();
    for (int s = 128; s > 0; s >>= 1) {
        if (threadIdx.x < s) red[threadIdx.x] = fmaxf(red[threadIdx.x], red[threadIdx.x + s]);
        __syncthreads();
    }
    mx = red[0]; __syncthreads();

    float sum = 0.0f;
#pragma unroll
    for (int i = 0; i < 4; ++i) { v[i] = __expf(v[i] - mx); sum += v[i]; }
    red[threadIdx.x] = sum; __syncthreads();
    for (int s = 128; s > 0; s >>= 1) {
        if (threadIdx.x < s) red[threadIdx.x] += red[threadIdx.x + s];
        __syncthreads();
    }
    const float inv = 1.0f / red[0];
#pragma unroll
    for (int i = 0; i < 4; ++i) p[threadIdx.x + i * 256] = v[i] * inv;
}

// ---- softmax over rows of length 128 (attn_feature), in place -------------
__global__ __launch_bounds__(128)
void softmax_feat(float* af) {
    float* p = af + ((long long)blockIdx.y * DVc + blockIdx.x) * DVc;
    __shared__ float red[128];
    float x = p[threadIdx.x];
    red[threadIdx.x] = x; __syncthreads();
    for (int s = 64; s > 0; s >>= 1) {
        if (threadIdx.x < s) red[threadIdx.x] = fmaxf(red[threadIdx.x], red[threadIdx.x + s]);
        __syncthreads();
    }
    const float mx = red[0]; __syncthreads();
    const float e = __expf(x - mx);
    red[threadIdx.x] = e; __syncthreads();
    for (int s = 64; s > 0; s >>= 1) {
        if (threadIdx.x < s) red[threadIdx.x] += red[threadIdx.x + s];
        __syncthreads();
    }
    p[threadIdx.x] = e / red[0];
}

// ---- LayerNorm over rows of length 1024 -----------------------------------
__global__ __launch_bounds__(256)
void layernorm_row(const float* xin, const float* gamma, const float* beta,
                   float* out) {
    const float* x = xin + (long long)blockIdx.x * Dc;
    float*       o = out + (long long)blockIdx.x * Dc;
    __shared__ float red[256];

    float v[4], s = 0.0f;
#pragma unroll
    for (int i = 0; i < 4; ++i) { v[i] = x[threadIdx.x + i * 256]; s += v[i]; }
    red[threadIdx.x] = s; __syncthreads();
    for (int t = 128; t > 0; t >>= 1) {
        if (threadIdx.x < t) red[threadIdx.x] += red[threadIdx.x + t];
        __syncthreads();
    }
    const float mean = red[0] * (1.0f / Dc); __syncthreads();

    float sq = 0.0f;
#pragma unroll
    for (int i = 0; i < 4; ++i) { const float d = v[i] - mean; sq += d * d; }
    red[threadIdx.x] = sq; __syncthreads();
    for (int t = 128; t > 0; t >>= 1) {
        if (threadIdx.x < t) red[threadIdx.x] += red[threadIdx.x + t];
        __syncthreads();
    }
    const float inv = rsqrtf(red[0] * (1.0f / Dc) + 1e-6f);
#pragma unroll
    for (int i = 0; i < 4; ++i) {
        const int c = threadIdx.x + i * 256;
        o[c] = (v[i] - mean) * inv * gamma[c] + beta[c];
    }
}

// ---------------------------------------------------------------------------
extern "C" void kernel_launch(void* const* d_in, const int* in_sizes, int n_in,
                              void* d_out, int out_size, void* d_ws, size_t ws_size,
                              hipStream_t stream) {
    (void)in_sizes; (void)n_in; (void)out_size; (void)ws_size;

    const float* q_time = (const float*)d_in[0];
    const float* k_time = (const float*)d_in[1];
    const float* q_feat = (const float*)d_in[2];
    const float* k_feat = (const float*)d_in[3];
    const float* v_in   = (const float*)d_in[4];
    const int*   mask   = (const int*)  d_in[5];
    const float* w_qt   = (const float*)d_in[6];
    const float* w_kt   = (const float*)d_in[7];
    const float* w_qf   = (const float*)d_in[8];
    const float* w_kf   = (const float*)d_in[9];
    const float* w_vs   = (const float*)d_in[10];
    const float* w_fc   = (const float*)d_in[11];
    const float* gamma  = (const float*)d_in[12];
    const float* beta   = (const float*)d_in[13];

    // d_out = [out(4M) | attn_time(32M) | attn_feature(0.5M)] floats
    float* outp = (float*)d_out;
    float* at   = outp + (long long)Bc * Nc * Dc;            // [B,H,N,N]
    float* af   = at   + (long long)Bc * Hc * Nc * Nc;       // [B,H,DV,DV]

    // workspace: 5 f16 projections + O1 + O2 (f16) + pre-LN f32 (~72 MB)
    const long long PROJ = (long long)Bc * Nc * Dc;          // 4194304 elems
    char* wsb = (char*)d_ws;
    void* qt16 = wsb + 0 * PROJ * 2;
    void* kt16 = wsb + 1 * PROJ * 2;
    void* qf16 = wsb + 2 * PROJ * 2;
    void* kf16 = wsb + 3 * PROJ * 2;
    void* vv16 = wsb + 4 * PROJ * 2;
    void* o1   = wsb + 5 * PROJ * 2;                         // [BH,1024,128] f16
    void* o2   = wsb + 6 * PROJ * 2;                         // [B,N,H*DV] f16
    float* lnin = (float*)(wsb + 7 * PROJ * 2);              // [B*N, D] f32

    const float inv_temp = 1.0f / sqrtf((float)DKc);
    const long long NN = (long long)Nc * Nc;

    GemmP p{};
    // ---- projections: [4096x1024] = X(f32) @ W^T(f32) -> f16 --------------
    const float* xs[5]  = { q_time, k_time, q_feat, k_feat, v_in };
    const float* ws5[5] = { w_qt, w_kt, w_qf, w_kf, w_vs };
    void* ps[5]         = { qt16, kt16, qf16, kf16, vv16 };
    for (int i = 0; i < 5; ++i) {
        p = GemmP{};
        p.A = xs[i];   p.lda_m = Dc; p.lda_k = 1;            // f32 k-fast
        p.Bm = ws5[i]; p.ldb_k = 1;  p.ldb_n = Dc;           // B(k,n)=W[n][k], f32 k-fast
        p.C = ps[i];   p.ldc = Hc * DKc;
        p.K = Dc; p.HH = 1; p.alpha = 1.0f;
        gemm_wmma<1, 1, 0><<<dim3(16, 32, 1), 256, 0, stream>>>(p);
    }

    // ---- S_time = (1/temp) * qt @ kt^T  -> f32 straight into d_out --------
    p = GemmP{};
    p.A = qt16; p.a_sb = (long long)Nc * Dc; p.a_sh = DKc;
    p.lda_m = Dc; p.lda_k = 1;                               // f16 k-fast
    p.Bm = kt16; p.b_sb = (long long)Nc * Dc; p.b_sh = DKc;
    p.ldb_k = 1; p.ldb_n = Dc;                               // B(d,m)=kt[m][d], f16 k-fast
    p.C = at; p.c_sb = (long long)Hc * NN; p.c_sh = NN; p.ldc = Nc;
    p.K = DKc; p.HH = Hc; p.alpha = inv_temp;
    gemm_wmma<0, 0, 1><<<dim3(16, 8, Bc * Hc), 256, 0, stream>>>(p);

    softmax_time<<<dim3(Nc, Bc * Hc, 1), 256, 0, stream>>>(at, mask);

    // ---- S_feat = (1/temp) * qf^T @ kf  -> f32 into d_out ------------------
    p = GemmP{};
    p.A = qf16; p.a_sb = (long long)Nc * Dc; p.a_sh = DVc;
    p.lda_m = 1; p.lda_k = Dc;                               // A(d,n)=qf[n][d], f16 m-fast
    p.Bm = kf16; p.b_sb = (long long)Nc * Dc; p.b_sh = DVc;
    p.ldb_k = Dc; p.ldb_n = 1;                               // B(n,e)=kf[n][e], f16 n-fast
    p.C = af; p.c_sb = (long long)Hc * DVc * DVc; p.c_sh = (long long)DVc * DVc;
    p.ldc = DVc;
    p.K = Nc; p.HH = Hc; p.alpha = inv_temp;
    gemm_wmma<2, 2, 1><<<dim3(2, 1, Bc * Hc), 256, 0, stream>>>(p);

    softmax_feat<<<dim3(DVc, Bc * Hc, 1), 128, 0, stream>>>(af);

    // ---- O1 = attn_time(f32, from d_out) @ vv16 -> f16 ---------------------
    p = GemmP{};
    p.A = at; p.a_sb = (long long)Hc * NN; p.a_sh = NN;
    p.lda_m = Nc; p.lda_k = 1;                               // f32 k-fast
    p.Bm = vv16; p.b_sb = (long long)Nc * Dc; p.b_sh = DVc;
    p.ldb_k = Dc; p.ldb_n = 1;                               // B(m,d)=vv[m][d], f16 n-fast
    p.C = o1; p.c_sb = (long long)Hc * Nc * DVc; p.c_sh = (long long)Nc * DVc;
    p.ldc = DVc;
    p.K = Nc; p.HH = Hc; p.alpha = 1.0f;
    gemm_wmma<1, 2, 0><<<dim3(2, 8, Bc * Hc), 256, 0, stream>>>(p);

    // ---- O2 = O1 @ attn_feature(f32, from d_out) -> f16 in [B,N,H*DV] ------
    p = GemmP{};
    p.A = o1; p.a_sb = (long long)Hc * Nc * DVc; p.a_sh = (long long)Nc * DVc;
    p.lda_m = DVc; p.lda_k = 1;                              // f16 k-fast
    p.Bm = af; p.b_sb = (long long)Hc * DVc * DVc; p.b_sh = (long long)DVc * DVc;
    p.ldb_k = DVc; p.ldb_n = 1;                              // B(d,e)=F[d][e], f32 n-fast
    p.C = o2; p.c_sb = (long long)Nc * Dc; p.c_sh = DVc; p.ldc = Hc * DVc;
    p.K = DVc; p.HH = Hc; p.alpha = 1.0f;
    gemm_wmma<0, 3, 0><<<dim3(2, 8, Bc * Hc), 256, 0, stream>>>(p);

    // ---- FC: lnin = O2 @ w_fc^T + v (residual fused) -----------------------
    p = GemmP{};
    p.A = o2; p.lda_m = Dc; p.lda_k = 1;                     // f16 k-fast
    p.Bm = w_fc; p.ldb_k = 1; p.ldb_n = Dc;                  // B(k,n)=w_fc[n][k], f32 k-fast
    p.C = lnin; p.ldc = Dc;
    p.aux = v_in;
    p.K = Dc; p.HH = 1; p.alpha = 1.0f;
    gemm_wmma<0, 1, 2><<<dim3(16, 32, 1), 256, 0, stream>>>(p);

    // ---- LayerNorm -> final output -----------------------------------------
    layernorm_row<<<dim3(Bc * Nc, 1, 1), 256, 0, stream>>>(lnin, gamma, beta, outp);
}
